// MultiHeadAttention_42528766165181
// MI455X (gfx1250) — matrix operand order repaired and checked by reading, not compiled, not run on hardware
//
#include <hip/hip_runtime.h>
#include <hip/hip_bf16.h>

typedef __attribute__((ext_vector_type(16))) __bf16 v16bf;
typedef __attribute__((ext_vector_type(8)))  float  v8f;

union ABf { v16bf v; unsigned u[8]; };
union PK  { __bf16 h[2]; unsigned u; };

constexpr int DIN = 1024, DOUT = 1024, NH = 16, HD = 64, SEQ = 2048, BATCH = 2;
constexpr int ROWS = BATCH * SEQ;

__device__ __forceinline__ unsigned pk2(float a, float b) {
    PK t; t.h[0] = (__bf16)a; t.h[1] = (__bf16)b; return t.u;
}
// A-matrix 16x32 bf16 layout (ISA 7.12.2): VGPR i holds K pair starting at:
__device__ __forceinline__ int kAoff(int i, int g) { return (i < 4 ? 2*i : 16 + 2*(i-4)) + 8*g; }
// B-matrix 32x16 bf16 layout: same K order as A (VGPR i holds K pair at kAoff)
__device__ __forceinline__ int kBoff(int i, int g) { return (i < 4 ? 2*i : 16 + 2*(i-4)) + 8*g; }

__device__ __forceinline__ float rsum16(float v) {
#pragma unroll
    for (int m = 1; m < 16; m <<= 1) v += __shfl_xor(v, m, 32);
    return v;
}
__device__ __forceinline__ float rmax16(float v) {
#pragma unroll
    for (int m = 1; m < 16; m <<= 1) v = fmaxf(v, __shfl_xor(v, m, 32));
    return v;
}

// Async copy of one bf16 pair (b32) from global into LDS, ASYNCcnt-tracked.
__device__ __forceinline__ void async_cp_b32(__bf16* lds_dst, const __bf16* gsrc) {
    unsigned lo = (unsigned)(size_t)(void*)lds_dst;        // flat addr[31:0] == LDS offset
    unsigned long long ga = (unsigned long long)(size_t)(const void*)gsrc;
    asm volatile("global_load_async_to_lds_b32 %0, %1, off" :: "v"(lo), "v"(ga) : "memory");
}
__device__ __forceinline__ void wait_async0() {
    asm volatile("s_wait_asynccnt 0x0" ::: "memory");
}

// ---------------------------------------------------------------------------
// Kernel 1: fused QKV projection + RMSNorm + RoPE (+ 1/sqrt(D) on Q)
// grid: (ROWS/256, 3*NH), block 256 (8 waves). Block tile: 256 x 64 (one head).
// Wave tile 32x64: 2 A-frags share 4 B-frags -> 8 WMMAs / K-step.
// ---------------------------------------------------------------------------
__global__ __launch_bounds__(256) void qkv_kernel(
    const float* __restrict__ x,  const float* __restrict__ Wq,
    const float* __restrict__ Wk, const float* __restrict__ Wv,
    const float* __restrict__ qs, const float* __restrict__ ks,
    __bf16* __restrict__ Qb, __bf16* __restrict__ Kb, __bf16* __restrict__ Vb)
{
    __shared__ __attribute__((aligned(16))) __bf16 As[256 * 34];
    __shared__ __attribute__((aligned(16))) __bf16 Bs[64 * 34];
    const int tid = threadIdx.x, lane = tid & 31, wave = tid >> 5;
    const int g = lane >> 4, n = lane & 15;
    const int bx = blockIdx.x;
    const int m  = blockIdx.y >> 4, h = blockIdx.y & 15;
    const float* W = (m == 0) ? Wq : (m == 1) ? Wk : Wv;
    const int n0 = h * 64;

    v8f acc[2][4] = {};
    for (int kk = 0; kk < DIN; kk += 32) {
        // stage A tile 256x32 (f32 -> bf16), row-major, stride 34
#pragma unroll
        for (int p = 0; p < 8; p++) {
            int row = p * 32 + (tid >> 3);
            int c   = (tid & 7) * 4;
            const float4 xv = *(const float4*)(x + (size_t)(bx * 256 + row) * DIN + kk + c);
            *(unsigned*)&As[row * 34 + c]     = pk2(xv.x, xv.y);
            *(unsigned*)&As[row * 34 + c + 2] = pk2(xv.z, xv.w);
        }
        // stage B tile 32x64 transposed -> Bs[n][k], stride 34
#pragma unroll
        for (int p = 0; p < 2; p++) {
            int kr = p * 16 + (tid >> 4);
            int c  = (tid & 15) * 4;
            const float4 wv = *(const float4*)(W + (size_t)(kk + kr) * DOUT + n0 + c);
            Bs[(c + 0) * 34 + kr] = (__bf16)wv.x;
            Bs[(c + 1) * 34 + kr] = (__bf16)wv.y;
            Bs[(c + 2) * 34 + kr] = (__bf16)wv.z;
            Bs[(c + 3) * 34 + kr] = (__bf16)wv.w;
        }
        __syncthreads();
        ABf a[2];
#pragma unroll
        for (int s = 0; s < 2; s++)
#pragma unroll
            for (int i = 0; i < 8; i++)
                a[s].u[i] = *(const unsigned*)&As[(wave * 32 + s * 16 + n) * 34 + kAoff(i, g)];
#pragma unroll
        for (int nf = 0; nf < 4; nf++) {
            ABf bfr;
#pragma unroll
            for (int i = 0; i < 8; i++)
                bfr.u[i] = *(const unsigned*)&Bs[(nf * 16 + n) * 34 + kBoff(i, g)];
            acc[0][nf] = __builtin_amdgcn_wmma_f32_16x16x32_bf16(
                false, a[0].v, false, bfr.v, (short)0, acc[0][nf], false, false);
            acc[1][nf] = __builtin_amdgcn_wmma_f32_16x16x32_bf16(
                false, a[1].v, false, bfr.v, (short)0, acc[1][nf], false, false);
        }
        __syncthreads();
    }

    __bf16* dst = (m == 0) ? Qb : (m == 1) ? Kb : Vb;
#pragma unroll
    for (int s = 0; s < 2; s++) {
        const int rowBase = bx * 256 + wave * 32 + s * 16;
        if (m <= 1) {
            const float* scl = (m == 0) ? qs : ks;
            const float sc0 = scl[n], sc1 = scl[16 + n], sc2 = scl[32 + n], sc3 = scl[48 + n];
            const float invf0 = __powf(10000.f, -((float)n)        * (1.f / 32.f));
            const float invf1 = __powf(10000.f, -((float)(16 + n)) * (1.f / 32.f));
            const float qsc = (m == 0) ? 0.125f : 1.f;   // 1/sqrt(64)
#pragma unroll
            for (int r = 0; r < 8; r++) {
                float ss = acc[s][0][r]*acc[s][0][r] + acc[s][1][r]*acc[s][1][r]
                         + acc[s][2][r]*acc[s][2][r] + acc[s][3][r]*acc[s][3][r];
                ss = rsum16(ss);
                const float rn = rsqrtf(ss * (1.f / 64.f) + 1e-6f);
                float t0 = acc[s][0][r]*rn*sc0, t1 = acc[s][1][r]*rn*sc1;
                float t2 = acc[s][2][r]*rn*sc2, t3 = acc[s][3][r]*rn*sc3;
                const int grow = rowBase + r + 8 * g;
                const int spos = grow & (SEQ - 1);
                float s0v, c0v, s1v, c1v;
                __sincosf(spos * invf0, &s0v, &c0v);
                __sincosf(spos * invf1, &s1v, &c1v);
                const float o0 = (t0 * c0v - t2 * s0v) * qsc;
                const float o1 = (t1 * c1v - t3 * s1v) * qsc;
                const float o2 = (t2 * c0v + t0 * s0v) * qsc;
                const float o3 = (t3 * c1v + t1 * s1v) * qsc;
                const int b = grow >> 11;
                size_t off = ((size_t)(b * NH + h) * SEQ + spos) * HD;
                dst[off + n]      = (__bf16)o0;
                dst[off + 16 + n] = (__bf16)o1;
                dst[off + 32 + n] = (__bf16)o2;
                dst[off + 48 + n] = (__bf16)o3;
            }
        } else {
#pragma unroll
            for (int r = 0; r < 8; r++) {
                const int grow = rowBase + r + 8 * g;
                const int b = grow >> 11, spos = grow & (SEQ - 1);
                size_t off = ((size_t)(b * NH + h) * SEQ + spos) * HD;
#pragma unroll
                for (int nf = 0; nf < 4; nf++)
                    dst[off + nf * 16 + n] = (__bf16)acc[s][nf][r];
            }
        }
    }
}

// ---------------------------------------------------------------------------
// Kernel 2: causal flash attention per (b,h,q-tile of 64). block 128 (4 waves).
// K/V tiles double-buffered and staged with GLOBAL_LOAD_ASYNC_TO_LDS_B32,
// tile t+1 issued before computing tile t (latency hidden behind WMMA work).
// ---------------------------------------------------------------------------
__global__ __launch_bounds__(128) void attn_kernel(
    const __bf16* __restrict__ Qb, const __bf16* __restrict__ Kb,
    const __bf16* __restrict__ Vb, __bf16* __restrict__ Ct)
{
    __shared__ __attribute__((aligned(16))) __bf16 Qs[64 * 66];
    __shared__ __attribute__((aligned(16))) __bf16 Kt[2][32 * 66];
    __shared__ __attribute__((aligned(16))) __bf16 Vr[2][32 * 66];
    __shared__ __attribute__((aligned(16))) __bf16 Pb[4][16 * 34];
    const int tid = threadIdx.x, lane = tid & 31, w = tid >> 5;
    const int g = lane >> 4, n = lane & 15;
    const int q0 = blockIdx.x * 64;
    const int b = blockIdx.y >> 4, h = blockIdx.y & 15;
    const size_t headOff = (size_t)(b * NH + h) * SEQ * HD;

    // ---- prologue: async-stage Q tile and K/V tile 0 ----
    {
        const __bf16* Qg = Qb + headOff + (size_t)q0 * HD;
#pragma unroll
        for (int j = 0; j < 16; j++) {
            int c = tid + j * 128;
            int row = (2 * c) >> 6, col = (2 * c) & 63;
            async_cp_b32(&Qs[row * 66 + col], Qg + 2 * c);
        }
        const __bf16* Kg = Kb + headOff;
        const __bf16* Vg = Vb + headOff;
#pragma unroll
        for (int j = 0; j < 8; j++) {
            int c = tid + j * 128;
            int row = (2 * c) >> 6, col = (2 * c) & 63;
            async_cp_b32(&Kt[0][row * 66 + col], Kg + 2 * c);
            async_cp_b32(&Vr[0][row * 66 + col], Vg + 2 * c);
        }
    }
    wait_async0();
    __syncthreads();

    ABf aq[2];
#pragma unroll
    for (int d0 = 0; d0 < 2; d0++)
#pragma unroll
        for (int i = 0; i < 8; i++)
            aq[d0].u[i] = *(const unsigned*)&Qs[(w * 16 + n) * 66 + d0 * 32 + kAoff(i, g)];

    float mrow[8], lrow[8];
    v8f O[4] = {};
#pragma unroll
    for (int r = 0; r < 8; r++) { mrow[r] = -1e30f; lrow[r] = 0.f; }

    const int ntiles = q0 / 32 + 2;
    for (int t = 0; t < ntiles; t++) {
        const int kv0 = t * 32;
        const int cur = t & 1;
        // issue async staging for tile t+1 into the other buffer
        if (t + 1 < ntiles) {
            const __bf16* Kg = Kb + headOff + (size_t)(kv0 + 32) * HD;
            const __bf16* Vg = Vb + headOff + (size_t)(kv0 + 32) * HD;
#pragma unroll
            for (int j = 0; j < 8; j++) {
                int c = tid + j * 128;
                int row = (2 * c) >> 6, col = (2 * c) & 63;
                async_cp_b32(&Kt[1 - cur][row * 66 + col], Kg + 2 * c);
                async_cp_b32(&Vr[1 - cur][row * 66 + col], Vg + 2 * c);
            }
        }
        // scores: S = Q(16x64) * K^T, accumulated over two d-halves
        v8f sc[2];
#pragma unroll
        for (int ns = 0; ns < 2; ns++) {
            v8f z = {};
#pragma unroll
            for (int dh = 0; dh < 2; dh++) {
                ABf bk;
#pragma unroll
                for (int i = 0; i < 8; i++)
                    bk.u[i] = *(const unsigned*)&Kt[cur][(ns * 16 + n) * 66 + dh * 32 + kBoff(i, g)];
                z = __builtin_amdgcn_wmma_f32_16x16x32_bf16(
                    false, aq[dh].v, false, bk.v, (short)0, z, false, false);
            }
            sc[ns] = z;
        }
        // causal mask (only boundary tiles for this wave need it)
        if (kv0 + 31 > q0 + w * 16) {
#pragma unroll
            for (int ns = 0; ns < 2; ns++)
#pragma unroll
                for (int r = 0; r < 8; r++) {
                    int kvp = kv0 + ns * 16 + n;
                    int qp  = q0 + w * 16 + r + 8 * g;
                    if (kvp > qp) sc[ns][r] = -1e30f;
                }
        }
        // online softmax + repack P into A-fragment layout via wave-private LDS
#pragma unroll
        for (int r = 0; r < 8; r++) {
            float mx   = rmax16(fmaxf(sc[0][r], sc[1][r]));
            float mnew = fmaxf(mrow[r], mx);
            float corr = __expf(mrow[r] - mnew);
            mrow[r] = mnew;
            float p0 = __expf(sc[0][r] - mnew);
            float p1 = __expf(sc[1][r] - mnew);
            lrow[r] = lrow[r] * corr + rsum16(p0 + p1);
#pragma unroll
            for (int f = 0; f < 4; f++) O[f][r] *= corr;
            Pb[w][(r + 8 * g) * 34 + n]      = (__bf16)p0;
            Pb[w][(r + 8 * g) * 34 + 16 + n] = (__bf16)p1;
        }
        // O += P(16x32) * V(32x64); V is row-major -> gather b16 pairs
        ABf ap;
#pragma unroll
        for (int i = 0; i < 8; i++)
            ap.u[i] = *(const unsigned*)&Pb[w][n * 34 + kAoff(i, g)];
#pragma unroll
        for (int f = 0; f < 4; f++) {
            ABf bv;
            const int dcol = f * 16 + n;
#pragma unroll
            for (int i = 0; i < 8; i++) {
                PK pp;
                pp.h[0] = Vr[cur][(kBoff(i, g) + 0) * 66 + dcol];
                pp.h[1] = Vr[cur][(kBoff(i, g) + 1) * 66 + dcol];
                bv.u[i] = pp.u;
            }
            O[f] = __builtin_amdgcn_wmma_f32_16x16x32_bf16(
                false, ap.v, false, bv.v, (short)0, O[f], false, false);
        }
        // own async copies done before the barrier -> next iter sees full tile
        wait_async0();
        __syncthreads();
    }
    // write ctx as bf16 in [B*S, H*D]
#pragma unroll
    for (int r = 0; r < 8; r++) {
        const float invl = 1.f / lrow[r];
        const size_t row = (size_t)(b * SEQ + q0 + w * 16 + r + 8 * g);
#pragma unroll
        for (int f = 0; f < 4; f++)
            Ct[row * (NH * HD) + h * HD + f * 16 + n] = (__bf16)(O[f][r] * invl);
    }
}

// ---------------------------------------------------------------------------
// Kernel 3: output projection ctx(4096x1024 bf16) @ W_out -> f32
// Block tile 256x64, wave tile 32x64 (8 WMMAs / K-step).
// ---------------------------------------------------------------------------
__global__ __launch_bounds__(256) void oproj_kernel(
    const __bf16* __restrict__ Ct, const float* __restrict__ Wo, float* __restrict__ out)
{
    __shared__ __attribute__((aligned(16))) __bf16 As[256 * 34];
    __shared__ __attribute__((aligned(16))) __bf16 Bs[64 * 34];
    const int tid = threadIdx.x, lane = tid & 31, wave = tid >> 5;
    const int g = lane >> 4, n = lane & 15;
    const int bx = blockIdx.x, n0 = blockIdx.y * 64;
    const unsigned* Ag = (const unsigned*)Ct;

    v8f acc[2][4] = {};
    for (int kk = 0; kk < DOUT; kk += 32) {
#pragma unroll
        for (int j = 0; j < 16; j++) {
            int c = tid + j * 256;
            int row = (2 * c) >> 5, col = (2 * c) & 31;
            *(unsigned*)&As[row * 34 + col] =
                Ag[((size_t)(bx * 256 + row) * DOUT + kk + col) >> 1];
        }
#pragma unroll
        for (int p = 0; p < 2; p++) {
            int kr = p * 16 + (tid >> 4);
            int c  = (tid & 15) * 4;
            const float4 wv = *(const float4*)(Wo + (size_t)(kk + kr) * DIN + n0 + c);
            Bs[(c + 0) * 34 + kr] = (__bf16)wv.x;
            Bs[(c + 1) * 34 + kr] = (__bf16)wv.y;
            Bs[(c + 2) * 34 + kr] = (__bf16)wv.z;
            Bs[(c + 3) * 34 + kr] = (__bf16)wv.w;
        }
        __syncthreads();
        ABf a[2];
#pragma unroll
        for (int s = 0; s < 2; s++)
#pragma unroll
            for (int i = 0; i < 8; i++)
                a[s].u[i] = *(const unsigned*)&As[(wave * 32 + s * 16 + n) * 34 + kAoff(i, g)];
#pragma unroll
        for (int nf = 0; nf < 4; nf++) {
            ABf bfr;
#pragma unroll
            for (int i = 0; i < 8; i++)
                bfr.u[i] = *(const unsigned*)&Bs[(nf * 16 + n) * 34 + kBoff(i, g)];
            acc[0][nf] = __builtin_amdgcn_wmma_f32_16x16x32_bf16(
                false, a[0].v, false, bfr.v, (short)0, acc[0][nf], false, false);
            acc[1][nf] = __builtin_amdgcn_wmma_f32_16x16x32_bf16(
                false, a[1].v, false, bfr.v, (short)0, acc[1][nf], false, false);
        }
        __syncthreads();
    }
#pragma unroll
    for (int s = 0; s < 2; s++)
#pragma unroll
        for (int r = 0; r < 8; r++) {
            const size_t row = (size_t)(bx * 256 + wave * 32 + s * 16 + r + 8 * g);
#pragma unroll
            for (int nf = 0; nf < 4; nf++)
                out[row * DIN + n0 + nf * 16 + n] = acc[s][nf][r];
        }
}

extern "C" void kernel_launch(void* const* d_in, const int* in_sizes, int n_in,
                              void* d_out, int out_size, void* d_ws, size_t ws_size,
                              hipStream_t stream)
{
    const float* x  = (const float*)d_in[0];
    const float* Wq = (const float*)d_in[1];
    const float* Wk = (const float*)d_in[2];
    const float* Wv = (const float*)d_in[3];
    const float* Wo = (const float*)d_in[4];
    const float* qs = (const float*)d_in[5];
    const float* ks = (const float*)d_in[6];
    float* out = (float*)d_out;

    const size_t perMat = (size_t)BATCH * NH * SEQ * HD;   // 4 Mi elements
    __bf16* Qb = (__bf16*)d_ws;
    __bf16* Kb = Qb + perMat;
    __bf16* Vb = Kb + perMat;
    __bf16* Ct = Vb + perMat;

    qkv_kernel<<<dim3(ROWS / 256, 3 * NH), 256, 0, stream>>>(x, Wq, Wk, Wv, qs, ks, Qb, Kb, Vb);
    attn_kernel<<<dim3(SEQ / 64, BATCH * NH), 128, 0, stream>>>(Qb, Kb, Vb, Ct);
    oproj_kernel<<<dim3(ROWS / 256, DIN / 64), 256, 0, stream>>>(Ct, Wo, out);
}
